// Pairwise_25434796327698
// MI455X (gfx1250) — matrix operand order repaired and checked by reading, not compiled, not run on hardware
//
#include <hip/hip_runtime.h>
#include <math.h>

typedef __attribute__((ext_vector_type(16))) _Float16 v16h;
typedef __attribute__((ext_vector_type(8)))  _Float16 v8h;
typedef __attribute__((ext_vector_type(8)))  float    v8f;

#define BTOT   256      // both towers folded into batch dim (2*128)
#define NNODES 1024
#define NTRIP  3069     // 3*(1024-1)

// ---------------------------------------------------------------------------
// Weight pack: w_conv [O, C, 3] f32 -> f16 row-major [O][K], K = 3*C
// ---------------------------------------------------------------------------
__global__ void cvt_f32_f16(const float* __restrict__ src,
                            _Float16* __restrict__ dst, int n) {
  int i = blockIdx.x * blockDim.x + threadIdx.x;
  if (i < n) dst[i] = (_Float16)src[i];
}

// ---------------------------------------------------------------------------
// Fused gather + (optional prev-layer layernorm + leakyReLU) + WMMA GEMM.
//   mode 0: source = raw f32 input (data1/data2), no normalization
//   mode 1: source = previous layer's raw f16 output + per-batch (mean,rstd),
//           applies (v-mean)*rstd then leaky(0.01) during the gather
// One workgroup (4 wave32) per (32-row M-tile, batch). Each wave owns a
// 16-wide O slice; per K-step it loads ONE B fragment and runs TWO
// v_wmma_f32_16x16x32_f16 against the two 16-row A sub-tiles -> B traffic
// halved, two independent accumulation chains.
// Output stored as raw pre-norm f16 at node index m+1 (node 0 left untouched).
// ---------------------------------------------------------------------------
__global__ __launch_bounds__(128)
void gemm_gather_wmma(int mode,
                      const float* __restrict__ data1,
                      const float* __restrict__ data2,
                      const _Float16* __restrict__ prev,
                      const float* __restrict__ stats,
                      const int* __restrict__ idx1,
                      const int* __restrict__ idx2,
                      const _Float16* __restrict__ wp,
                      const float* __restrict__ bias,
                      _Float16* __restrict__ yout,
                      int Cin, int K, int Ofull)
{
  __shared__ __align__(16) _Float16 At[32 * 768];   // max K = 768 -> 48 KB
  __shared__ int cols[96];                          // 32 rows x 3 gather cols

  const int b  = blockIdx.y;
  const int m0 = blockIdx.x * 32;                   // first gathered row
  const int* idx = (b < 128) ? (idx1 + (size_t)b * NTRIP)
                             : (idx2 + (size_t)(b - 128) * NTRIP);

  // stage the 96 contiguous triple-indices for this tile
  if (threadIdx.x < 96) {
    const int t = threadIdx.x;
    cols[t] = (3 * m0 + t < 3 * 1023) ? idx[3 * m0 + t] : 0;
  }

  float mean = 0.f, rstd = 1.f;
  if (mode) { mean = stats[2 * b]; rstd = stats[2 * b + 1]; }
  __syncthreads();

  // cooperative gather of the 32 x K activation tile into LDS (f16)
  for (int r = 0; r < 32; ++r) {
    const int m = m0 + r;
    _Float16* dst = &At[r * K];
    if (m < 1023) {
      for (int kk = threadIdx.x; kk < K; kk += 128) {
        const int c   = kk / 3;
        const int k3  = kk - 3 * c;
        const int col = cols[3 * r + k3];
        float v;
        if (mode == 0) {
          const float* src = (b < 128)
              ? (data1 + ((size_t)b * Cin + c) * NNODES)
              : (data2 + ((size_t)(b - 128) * Cin + c) * NNODES);
          v = src[col];
        } else {
          v = (float)prev[((size_t)b * Cin + c) * NNODES + col];
          v = (v - mean) * rstd;               // fused tensor-layernorm
          v = v > 0.f ? v : 0.01f * v;         // fused leaky ReLU
        }
        dst[kk] = (_Float16)v;
      }
    } else {
      for (int kk = threadIdx.x; kk < K; kk += 128) dst[kk] = (_Float16)0.f;
    }
  }
  __syncthreads();

  const int lane = threadIdx.x & 31;
  const int wv   = threadIdx.x >> 5;
  const int l15  = lane & 15;
  const int hh   = lane >> 4;     // which 16-lane half of the wave
  const int nkb  = K >> 5;        // K is always a multiple of 32

  for (int ot = wv; ot < (Ofull >> 4); ot += 4) {
    v8f acc0 = {}, acc1 = {};
    const int o = ot * 16 + l15;
    for (int kb = 0; kb < nkb; ++kb) {
      union { v16h v; v8h h[2]; } ua0, ua1, ub;
      // B frag (32x16 f16): lane holds col N=l15, contiguous 16 halfs per half-wave
      const v8h* bp = (const v8h*)&wp[(size_t)o * K + kb * 32 + hh * 16];
      ub.h[0] = bp[0];
      ub.h[1] = bp[1];
      // A frags (16x32 f16): lane<16 -> K {0..15}, lane>=16 -> {16..31}
      const v8h* ap0 = (const v8h*)&At[l15 * K + kb * 32 + hh * 16];
      ua0.h[0] = ap0[0];
      ua0.h[1] = ap0[1];          // +8 halfs
      const v8h* ap1 = (const v8h*)&At[(16 + l15) * K + kb * 32 + hh * 16];
      ua1.h[0] = ap1[0];
      ua1.h[1] = ap1[1];
      acc0 = __builtin_amdgcn_wmma_f32_16x16x32_f16(
          false, ua0.v, false, ub.v, (short)0, acc0, false, false);
      acc1 = __builtin_amdgcn_wmma_f32_16x16x32_f16(
          false, ua1.v, false, ub.v, (short)0, acc1, false, false);
    }
    const float bs = bias[o];
    _Float16* yo = yout + ((size_t)b * Ofull + o) * NNODES;
    const int mb0 = m0 + hh * 8;        // D: VGPR v -> M = v + 8*hh
    const int mb1 = m0 + 16 + hh * 8;
#pragma unroll
    for (int v = 0; v < 8; ++v) {
      const int ma = mb0 + v;
      if (ma < 1023) yo[ma + 1] = (_Float16)(acc0[v] + bs);
      const int mbv = mb1 + v;
      if (mbv < 1023) yo[mbv + 1] = (_Float16)(acc1[v] + bs);
    }
  }
}

// ---------------------------------------------------------------------------
// Per-batch layernorm stats over [O, 1024] (node 0 counted as exact zero,
// unbiased variance like torch.std). One 256-thread block per batch element.
// ---------------------------------------------------------------------------
__global__ __launch_bounds__(256)
void stats_kernel(const _Float16* __restrict__ Y, float* __restrict__ st, int O) {
  __shared__ float ss[256], sq[256];
  const int b = blockIdx.x;
  float s = 0.f, q = 0.f;
  for (int o = 0; o < O; ++o) {
    const _Float16* row = Y + ((size_t)b * O + o) * NNODES;
    for (int n = 1 + threadIdx.x; n < NNODES; n += 256) {
      const float v = (float)row[n];
      s += v; q += v * v;
    }
  }
  ss[threadIdx.x] = s; sq[threadIdx.x] = q;
  __syncthreads();
  for (int d = 128; d > 0; d >>= 1) {
    if (threadIdx.x < d) {
      ss[threadIdx.x] += ss[threadIdx.x + d];
      sq[threadIdx.x] += sq[threadIdx.x + d];
    }
    __syncthreads();
  }
  if (threadIdx.x == 0) {
    const float cnt  = (float)O * NNODES;     // zero-node included in count
    const float mean = ss[0] / cnt;
    const float var  = (sq[0] - ss[0] * ss[0] / cnt) / (cnt - 1.f);
    const float rstd = 1.f / (sqrtf(fmaxf(var, 0.f)) + 1e-5f);
    st[2 * b]     = mean;
    st[2 * b + 1] = rstd;
  }
}

// ---------------------------------------------------------------------------
// Max over nodes with layer-3 norm fused: rstd>0 so
// max_n (v-mean)*rstd == (max_n v - mean)*rstd, max includes node0's v=0.
// One wave32 per (b, o); 4 waves per block.
// ---------------------------------------------------------------------------
__global__ __launch_bounds__(128)
void maxpool_kernel(const _Float16* __restrict__ Y,
                    const float* __restrict__ st,
                    float* __restrict__ pooled) {
  const int b    = blockIdx.y;
  const int o    = blockIdx.x * 4 + (threadIdx.x >> 5);
  const int lane = threadIdx.x & 31;
  const _Float16* row = Y + ((size_t)b * 64 + o) * NNODES;
  float mv = 0.f;                                  // node 0 raw value
  for (int n = 1 + lane; n < NNODES; n += 32) mv = fmaxf(mv, (float)row[n]);
  for (int d = 16; d > 0; d >>= 1) mv = fmaxf(mv, __shfl_xor(mv, d, 32));
  if (lane == 0) pooled[b * 64 + o] = (mv - st[2 * b]) * st[2 * b + 1];
}

// ---------------------------------------------------------------------------
// FC head for both towers + sigmoid(y1-y2). Thread b handles batch row b.
// ---------------------------------------------------------------------------
__global__ void fc_kernel(const float* __restrict__ pooled,
                          const float* __restrict__ wfc1,
                          const float* __restrict__ bfc1,
                          const float* __restrict__ wfc2,
                          const float* __restrict__ bfc2,
                          float* __restrict__ out) {
  const int b = threadIdx.x;
  if (b >= 128) return;
  float s[2];
#pragma unroll
  for (int t = 0; t < 2; ++t) {
    const float* x = pooled + (size_t)(t * 128 + b) * 64;
    float acc2 = bfc2[0];
    for (int j = 0; j < 32; ++j) {
      float h = bfc1[j];
      const float* wr = wfc1 + j * 64;
      for (int c = 0; c < 64; ++c) h += x[c] * wr[c];
      h = h > 0.f ? h : 0.01f * h;
      acc2 += h * wfc2[j];
    }
    s[t] = acc2;
  }
  out[b] = 1.f / (1.f + expf(-(s[0] - s[1])));
}

// ---------------------------------------------------------------------------
extern "C" void kernel_launch(void* const* d_in, const int* in_sizes, int n_in,
                              void* d_out, int out_size, void* d_ws, size_t ws_size,
                              hipStream_t stream) {
  (void)in_sizes; (void)n_in; (void)out_size; (void)ws_size;

  const float* data1 = (const float*)d_in[0];
  const int*   idx1  = (const int*)d_in[1];
  const float* data2 = (const float*)d_in[2];
  const int*   idx2  = (const int*)d_in[3];
  const float* w1  = (const float*)d_in[4];
  const float* bb1 = (const float*)d_in[5];
  const float* w2  = (const float*)d_in[6];
  const float* bb2 = (const float*)d_in[7];
  const float* w3  = (const float*)d_in[8];
  const float* bb3 = (const float*)d_in[9];
  const float* wfc1 = (const float*)d_in[10];
  const float* bfc1 = (const float*)d_in[11];
  const float* wfc2 = (const float*)d_in[12];
  const float* bfc2 = (const float*)d_in[13];
  float* out = (float*)d_out;

  char* ws = (char*)d_ws;
  size_t off = 0;
  auto take = [&](size_t bytes) -> char* {
    char* p = ws + off;
    off = (off + bytes + 255) & ~(size_t)255;
    return p;
  };
  _Float16* wp1    = (_Float16*)take((size_t)256 * 192 * 2);
  _Float16* wp2    = (_Float16*)take((size_t)128 * 768 * 2);
  _Float16* wp3    = (_Float16*)take((size_t)64 * 384 * 2);
  float*    st     = (float*)take((size_t)3 * BTOT * 2 * 4);
  float*    pooled = (float*)take((size_t)BTOT * 64 * 4);
  _Float16* Y2     = (_Float16*)take((size_t)BTOT * 128 * NNODES * 2);
  _Float16* Y1     = (_Float16*)take((size_t)BTOT * 256 * NNODES * 2);
  _Float16* Y3     = Y1;               // alias: Y1 is dead once layer 3 runs
  float* st1 = st, *st2 = st + 2 * BTOT, *st3 = st + 4 * BTOT;

  // pack conv weights to f16 [O][K]
  cvt_f32_f16<<<(256 * 192 + 255) / 256, 256, 0, stream>>>(w1, wp1, 256 * 192);
  cvt_f32_f16<<<(128 * 768 + 255) / 256, 256, 0, stream>>>(w2, wp2, 128 * 768);
  cvt_f32_f16<<<(64 * 384 + 255) / 256, 256, 0, stream>>>(w3, wp3, 64 * 384);

  const dim3 ggrid(32, BTOT);   // 32 M-tiles (32 rows each) x (2 towers * 128)

  // layer 1: gather raw f32 input, GEMM K=192 -> O=256
  gemm_gather_wmma<<<ggrid, 128, 0, stream>>>(0, data1, data2, nullptr, nullptr,
                                              idx1, idx2, wp1, bb1, Y1, 64, 192, 256);
  stats_kernel<<<BTOT, 256, 0, stream>>>(Y1, st1, 256);

  // layer 2: gather Y1 with fused norm+leaky, GEMM K=768 -> O=128
  gemm_gather_wmma<<<ggrid, 128, 0, stream>>>(1, nullptr, nullptr, Y1, st1,
                                              idx1, idx2, wp2, bb2, Y2, 256, 768, 128);
  stats_kernel<<<BTOT, 256, 0, stream>>>(Y2, st2, 128);

  // layer 3: gather Y2 with fused norm+leaky, GEMM K=384 -> O=64
  gemm_gather_wmma<<<ggrid, 128, 0, stream>>>(1, nullptr, nullptr, Y2, st2,
                                              idx1, idx2, wp3, bb3, Y3, 128, 384, 64);
  stats_kernel<<<BTOT, 256, 0, stream>>>(Y3, st3, 64);

  // max-pool with layer-3 norm fused, then FC head + sigmoid(y1-y2)
  maxpool_kernel<<<dim3(16, BTOT), 128, 0, stream>>>(Y3, st3, pooled);
  fc_kernel<<<1, 128, 0, stream>>>(pooled, wfc1, bfc1, wfc2, bfc2, out);
}
